// GraphNetworkSimulator_45672682226304
// MI455X (gfx1250) — matrix operand order repaired and checked by reading, not compiled, not run on hardware
//
#include <hip/hip_runtime.h>

// ---------------------------------------------------------------------------
// CDNA5 (gfx1250, wave32) fused GraphNet implementation, v2.
//  - v_wmma_f32_16x16x32_f16 for all GEMMs (f16 in, f32 accumulate)
//  - 32 rows/block: each wave does 2 row-tiles per B-fragment load
//  - f16 mirror of h_n; gathered rows staged via global_load_async_to_lds_b128
// ---------------------------------------------------------------------------

typedef __attribute__((ext_vector_type(16))) _Float16     v16h;
typedef __attribute__((ext_vector_type(8)))  float        v8f;
typedef __attribute__((ext_vector_type(4)))  unsigned int v4u;

#define WMMA_F16(a, b, c) \
  __builtin_amdgcn_wmma_f32_16x16x32_f16(false, (a), false, (b), (short)0, (c), false, false)

enum { M_ENC_NODE = 0, M_ENC_EDGE, M_EDGE_STEP, M_NODE_STEP, M_DEC_NODE, M_DEC_EDGE };

// Async copy 16 bytes global -> LDS (gfx1250 VGLOBAL op, tracked by ASYNCcnt).
__device__ __forceinline__ void async_copy_b128(unsigned ldsByte, unsigned long long gAddr) {
  asm volatile("global_load_async_to_lds_b128 %0, %1, off"
               :: "v"(ldsByte), "v"(gAddr) : "memory");
}
__device__ __forceinline__ void async_wait0() {
  asm volatile("s_wait_asynccnt 0x0" ::: "memory");
}

// Load a 16x32 f16 A-fragment for one lane from an LDS row (row-major, rowPtr
// points at k=0 of row m = lane&15). ISA layout (CDNA5 ISA, 16-bit A 16x32):
//   lanes 0-15 : e in [0,8) -> k = e      ; e in [8,16) -> k = 16 + (e-8)
//   lanes 16-31: e in [0,8) -> k = 8 + e  ; e in [8,16) -> k = 24 + (e-8)
// Both 8-element runs are 16B aligned -> two ds_load_b128.
__device__ __forceinline__ v16h load_a_frag(const _Float16* __restrict__ rowPtr,
                                            int kt, int half) {
  union { v16h h; v4u u[2]; } A;
  const _Float16* p = rowPtr + kt * 32 + half * 8;
  A.u[0] = *(const v4u*)(p);
  A.u[1] = *(const v4u*)(p + 16);
  return A.h;
}

// ---------------------------------------------------------------------------
// Weight pre-pack: f32 [Ksrc, Nsrc] row-major -> f16 WMMA-B lane layout.
// B (32x16 f16, KxN): lane = half*16 + n_local holds column n; element e holds
// k = half*8 + e (e<8), k = 16 + half*8 + (e-8) (e>=8). Packed flat: P[((nt*KT + kt)*32 + lane)*16 + e] so each
// lane's fragment is one contiguous 32B-aligned 32-byte load.
// ---------------------------------------------------------------------------
__global__ void pack_w(const float* __restrict__ W, _Float16* __restrict__ P,
                       int Ksrc, int Nsrc, int KT, int NT) {
  int t = blockIdx.x * 256 + threadIdx.x;
  int total = NT * KT * 512;
  if (t >= total) return;
  int e     = t & 15;
  int lane  = (t >> 4) & 31;
  int chunk = t >> 9;
  int kt = chunk % KT, nt = chunk / KT;
  int half = lane >> 4, nl = lane & 15;
  int k = kt * 32 + half * 8 + (e & 7) + ((e >> 3) << 4);
  int n = nt * 16 + nl;
  float v = (k < Ksrc && n < Nsrc) ? W[(size_t)k * Nsrc + n] : 0.0f;
  P[t] = (_Float16)v;
}

__global__ void zero_f32(float* __restrict__ p, long long n) {
  long long i = (long long)blockIdx.x * 256 + threadIdx.x;
  if (i < n) p[i] = 0.0f;
}

// ---------------------------------------------------------------------------
// Generic fused 2-layer MLP over 32 rows/block, 256 threads = 8 waves.
// Wave w owns output cols [16w,16w+16) of BOTH 16-row tiles (rt=0,1), so each
// B-fragment load feeds two v_wmma. A (32 x K1PAD) is built in LDS per MODE.
// ---------------------------------------------------------------------------
template <int K1PAD, int NT2, int MODE>
__global__ __launch_bounds__(256) void fused_mlp(
    const float* __restrict__ src0,            // primary f32 rows
    const float* __restrict__ src1,            // sent (NODE)
    const float* __restrict__ src2,            // recv (NODE)
    const _Float16* __restrict__ hn16,         // f16 mirror of h_n (EDGE gather / NODE rows)
    const int* __restrict__ senders, const int* __restrict__ receivers,
    const _Float16* __restrict__ W1p, const float* __restrict__ b1,
    const _Float16* __restrict__ W2p, const float* __restrict__ b2,
    float* __restrict__ dst,                   // h_n / h_e / out
    _Float16* __restrict__ dst16,              // f16 mirror to update (or null)
    float* __restrict__ accS, float* __restrict__ accR,
    int outW) {
  static_assert(K1PAD % 32 == 0, "K1PAD must be a multiple of 32");
  constexpr int KT1 = K1PAD / 32;

  __shared__ __align__(16) _Float16 lA[32 * K1PAD];
  __shared__ __align__(16) _Float16 lH[32 * 128];
  __shared__ int lS[32], lR[32];

  const int tid = threadIdx.x;
  const int r0  = blockIdx.x * 32;

  if constexpr (MODE == M_EDGE_STEP) {
    if (tid < 32) { lS[tid] = senders[r0 + tid]; lR[tid] = receivers[r0 + tid]; }
    __syncthreads();
  }

  const unsigned laBase = (unsigned)(unsigned long long)(&lA[0]);

  // ---- stage A tile (32 x K1PAD) into LDS as f16 ----
  if constexpr (MODE == M_EDGE_STEP) {
    // gathered h_n rows: async raw f16 copy, 32 rows x 2 gathers x 256B = 1024 b128s
    const unsigned long long hb = (unsigned long long)hn16;
#pragma unroll
    for (int i = 0; i < 4; ++i) {
      int t = tid + i * 256;
      int c = t & 15, r = (t >> 4) & 31, g = t >> 9;
      int idx = g ? lR[r] : lS[r];
      async_copy_b128(laBase + (unsigned)(r * 768 + 256 + g * 256 + c * 16),
                      hb + (unsigned long long)idx * 256u + (unsigned)(c * 16));
    }
    // own h_e rows: f32 -> f16 convert (k < 128)
    for (int t = tid; t < 32 * 128; t += 256) {
      int r = t >> 7, k = t & 127;
      lA[r * K1PAD + k] = (_Float16)src0[(size_t)(r0 + r) * 128 + k];
    }
    async_wait0();
  } else if constexpr (MODE == M_NODE_STEP) {
    // own h_n rows: async raw f16 copy, 32 rows x 256B = 512 b128s
    const unsigned long long hb = (unsigned long long)hn16;
#pragma unroll
    for (int i = 0; i < 2; ++i) {
      int t = tid + i * 256;
      int c = t & 15, r = t >> 4;
      async_copy_b128(laBase + (unsigned)(r * 768 + c * 16),
                      hb + (unsigned long long)(r0 + r) * 256u + (unsigned)(c * 16));
    }
    // sent | recv: f32 -> f16 convert (k in [128,384))
    for (int t = tid; t < 32 * 256; t += 256) {
      int r = t >> 8, kk = t & 255;
      float v = (kk < 128) ? src1[(size_t)(r0 + r) * 128 + kk]
                           : src2[(size_t)(r0 + r) * 128 + (kk - 128)];
      lA[r * K1PAD + 128 + kk] = (_Float16)v;
    }
    async_wait0();
  } else if constexpr (MODE == M_ENC_NODE) {
    for (int t = tid; t < 32 * K1PAD; t += 256) {
      int r = t / K1PAD, k = t - r * K1PAD;
      lA[t] = (_Float16)((k < 16) ? src0[(size_t)(r0 + r) * 16 + k] : 0.0f);
    }
  } else if constexpr (MODE == M_ENC_EDGE) {
    for (int t = tid; t < 32 * K1PAD; t += 256) {
      int r = t / K1PAD, k = t - r * K1PAD;
      lA[t] = (_Float16)((k < 8) ? src0[(size_t)(r0 + r) * 8 + k] : 0.0f);
    }
  } else {  // decoders: plain f32 [rows,128]
    for (int t = tid; t < 32 * 128; t += 256) {
      int r = t >> 7, k = t & 127;
      lA[t] = (_Float16)src0[(size_t)(r0 + r) * 128 + k];
    }
  }
  __syncthreads();

  const int w = tid >> 5, lane = tid & 31;
  const int half = lane >> 4, m = lane & 15;
  const int n = w * 16 + m;  // this lane's output column (C/D layout: N = lane&15)

  // ---- layer 1: hidden = relu(A @ W1 + b1), both row tiles ----
  v8f acc0 = {}, acc1 = {};
  {
    const _Float16* aRow0 = &lA[m * K1PAD];
    const _Float16* aRow1 = &lA[(16 + m) * K1PAD];
    const _Float16* Bp    = W1p + (size_t)(w * KT1) * 512 + lane * 16;
#pragma unroll
    for (int kt = 0; kt < KT1; ++kt) {
      v16h b  = *(const v16h*)(Bp + (size_t)kt * 512);
      v16h a0 = load_a_frag(aRow0, kt, half);
      v16h a1 = load_a_frag(aRow1, kt, half);
      acc0 = WMMA_F16(a0, b, acc0);
      acc1 = WMMA_F16(a1, b, acc1);
    }
  }
  {
    float bias = b1[n];
#pragma unroll
    for (int r = 0; r < 8; ++r) {  // C/D layout: row M = half*8 + r
      float v0 = acc0[r] + bias;
      float v1 = acc1[r] + bias;
      lH[(half * 8 + r) * 128 + n]        = (_Float16)(v0 > 0.0f ? v0 : 0.0f);
      lH[(16 + half * 8 + r) * 128 + n]   = (_Float16)(v1 > 0.0f ? v1 : 0.0f);
    }
  }
  __syncthreads();

  // ---- layer 2: out = hidden @ W2 + b2 ----
  if (w < NT2) {  // wave-uniform: EXEC stays all-ones inside
    v8f acc2a = {}, acc2b = {};
    const _Float16* hRow0 = &lH[m * 128];
    const _Float16* hRow1 = &lH[(16 + m) * 128];
    const _Float16* Bp    = W2p + (size_t)(w * 4) * 512 + lane * 16;
#pragma unroll
    for (int kt = 0; kt < 4; ++kt) {
      v16h b  = *(const v16h*)(Bp + (size_t)kt * 512);
      v16h a0 = load_a_frag(hRow0, kt, half);
      v16h a1 = load_a_frag(hRow1, kt, half);
      acc2a = WMMA_F16(a0, b, acc2a);
      acc2b = WMMA_F16(a1, b, acc2b);
    }

    if constexpr (MODE == M_DEC_NODE || MODE == M_DEC_EDGE) {
      if (n < outW) {
        float bias = b2[n];
#pragma unroll
        for (int r = 0; r < 8; ++r) {
          dst[(size_t)(r0 + half * 8 + r) * outW + n]      = acc2a[r] + bias;
          dst[(size_t)(r0 + 16 + half * 8 + r) * outW + n] = acc2b[r] + bias;
        }
      }
    } else if constexpr (MODE == M_EDGE_STEP) {
      float bias = b2[n];
#pragma unroll
      for (int rt = 0; rt < 2; ++rt) {
#pragma unroll
        for (int r = 0; r < 8; ++r) {
          int lr = rt * 16 + half * 8 + r;
          float e = (rt ? acc2b[r] : acc2a[r]) + bias;       // e_new
          dst[(size_t)(r0 + lr) * 128 + n] += e;             // h_e residual (row owned)
          atomicAdd(&accS[(size_t)lS[lr] * 128 + n], e);     // segment_sum senders
          atomicAdd(&accR[(size_t)lR[lr] * 128 + n], e);     // segment_sum receivers
        }
      }
    } else if constexpr (MODE == M_NODE_STEP) {
      float bias = b2[n];
#pragma unroll
      for (int rt = 0; rt < 2; ++rt) {
#pragma unroll
        for (int r = 0; r < 8; ++r) {
          size_t row = (size_t)(r0 + rt * 16 + half * 8 + r);
          float v = dst[row * 128 + n] + (rt ? acc2b[r] : acc2a[r]) + bias;  // residual
          dst[row * 128 + n]   = v;
          dst16[row * 128 + n] = (_Float16)v;                 // keep f16 mirror in sync
        }
      }
    } else if constexpr (MODE == M_ENC_NODE) {
      float bias = b2[n];
#pragma unroll
      for (int rt = 0; rt < 2; ++rt) {
#pragma unroll
        for (int r = 0; r < 8; ++r) {
          size_t row = (size_t)(r0 + rt * 16 + half * 8 + r);
          float v = (rt ? acc2b[r] : acc2a[r]) + bias;
          dst[row * 128 + n]   = v;
          dst16[row * 128 + n] = (_Float16)v;
        }
      }
    } else {  // M_ENC_EDGE: plain store
      float bias = b2[n];
#pragma unroll
      for (int r = 0; r < 8; ++r) {
        dst[(size_t)(r0 + half * 8 + r) * 128 + n]      = acc2a[r] + bias;
        dst[(size_t)(r0 + 16 + half * 8 + r) * 128 + n] = acc2b[r] + bias;
      }
    }
  }
}

// ---------------------------------------------------------------------------
// Host side
// ---------------------------------------------------------------------------
extern "C" void kernel_launch(void* const* d_in, const int* in_sizes, int n_in,
                              void* d_out, int out_size, void* d_ws, size_t ws_size,
                              hipStream_t stream) {
  constexpr int N = 20000, E = 160000, L = 128, S = 5, ON = 3;

  const float* nodes        = (const float*)d_in[0];
  const float* edges        = (const float*)d_in[1];
  const int*   senders      = (const int*)d_in[2];
  const int*   receivers    = (const int*)d_in[3];
  const float* enc_node_W1  = (const float*)d_in[4];
  const float* enc_node_b1  = (const float*)d_in[5];
  const float* enc_node_W2  = (const float*)d_in[6];
  const float* enc_node_b2  = (const float*)d_in[7];
  const float* enc_edge_W1  = (const float*)d_in[8];
  const float* enc_edge_b1  = (const float*)d_in[9];
  const float* enc_edge_W2  = (const float*)d_in[10];
  const float* enc_edge_b2  = (const float*)d_in[11];
  const float* eW1 = (const float*)d_in[12];
  const float* eb1 = (const float*)d_in[13];
  const float* eW2 = (const float*)d_in[14];
  const float* eb2 = (const float*)d_in[15];
  const float* nW1 = (const float*)d_in[16];
  const float* nb1 = (const float*)d_in[17];
  const float* nW2 = (const float*)d_in[18];
  const float* nb2 = (const float*)d_in[19];
  const float* dec_node_W1 = (const float*)d_in[20];
  const float* dec_node_b1 = (const float*)d_in[21];
  const float* dec_node_W2 = (const float*)d_in[22];
  const float* dec_node_b2 = (const float*)d_in[23];
  const float* dec_edge_W1 = (const float*)d_in[24];
  const float* dec_edge_b1 = (const float*)d_in[25];
  const float* dec_edge_W2 = (const float*)d_in[26];
  const float* dec_edge_b2 = (const float*)d_in[27];

  // ---- workspace carve-up (256B aligned) ----
  char*  ws  = (char*)d_ws;
  size_t off = 0;
  auto carve = [&](size_t bytes) -> void* {
    void* p = ws + off;
    off = (off + bytes + 255) & ~(size_t)255;
    return p;
  };
  float*     h_n   = (float*)carve((size_t)N * 128 * 4);
  float*     h_e   = (float*)carve((size_t)E * 128 * 4);
  float*     sent  = (float*)carve((size_t)2 * N * 128 * 4);  // sent | recv contiguous
  float*     recv  = sent + (size_t)N * 128;
  _Float16*  h_n16 = (_Float16*)carve((size_t)N * 128 * 2);   // f16 mirror of h_n

  auto carveH = [&](size_t elems) -> _Float16* { return (_Float16*)carve(elems * 2); };
  _Float16* P_encNW1 = carveH(8 * 1 * 512);
  _Float16* P_encNW2 = carveH(8 * 4 * 512);
  _Float16* P_encEW1 = carveH(8 * 1 * 512);
  _Float16* P_encEW2 = carveH(8 * 4 * 512);
  _Float16* P_eW1    = carveH((size_t)S * 8 * 12 * 512);
  _Float16* P_eW2    = carveH((size_t)S * 8 * 4 * 512);
  _Float16* P_nW1    = carveH((size_t)S * 8 * 12 * 512);
  _Float16* P_nW2    = carveH((size_t)S * 8 * 4 * 512);
  _Float16* P_dNW1   = carveH(8 * 4 * 512);
  _Float16* P_dNW2   = carveH(1 * 4 * 512);
  _Float16* P_dEW1   = carveH(8 * 4 * 512);
  _Float16* P_dEW2   = carveH(1 * 4 * 512);
  (void)ws_size; (void)n_in; (void)in_sizes; (void)out_size;

  auto pack = [&](const float* W, _Float16* P, int Ksrc, int Nsrc, int KT, int NT) {
    int total = NT * KT * 512;
    pack_w<<<(total + 255) / 256, 256, 0, stream>>>(W, P, Ksrc, Nsrc, KT, NT);
  };

  // ---- pre-pack all weights into WMMA-B layout ----
  pack(enc_node_W1, P_encNW1, 16, 128, 1, 8);
  pack(enc_node_W2, P_encNW2, 128, 128, 4, 8);
  pack(enc_edge_W1, P_encEW1, 8, 128, 1, 8);
  pack(enc_edge_W2, P_encEW2, 128, 128, 4, 8);
  for (int i = 0; i < S; ++i) {
    pack(eW1 + (size_t)i * 384 * 128, P_eW1 + (size_t)i * 8 * 12 * 512, 384, 128, 12, 8);
    pack(eW2 + (size_t)i * 128 * 128, P_eW2 + (size_t)i * 8 * 4 * 512, 128, 128, 4, 8);
    pack(nW1 + (size_t)i * 384 * 128, P_nW1 + (size_t)i * 8 * 12 * 512, 384, 128, 12, 8);
    pack(nW2 + (size_t)i * 128 * 128, P_nW2 + (size_t)i * 8 * 4 * 512, 128, 128, 4, 8);
  }
  pack(dec_node_W1, P_dNW1, 128, 128, 4, 8);
  pack(dec_node_W2, P_dNW2, 128, ON, 4, 1);
  pack(dec_edge_W1, P_dEW1, 128, 128, 4, 8);
  pack(dec_edge_W2, P_dEW2, 128, 1, 4, 1);

  // ---- encode ----
  fused_mlp<32, 8, M_ENC_NODE><<<N / 32, 256, 0, stream>>>(
      nodes, nullptr, nullptr, nullptr, nullptr, nullptr,
      P_encNW1, enc_node_b1, P_encNW2, enc_node_b2, h_n, h_n16, nullptr, nullptr, 128);
  fused_mlp<32, 8, M_ENC_EDGE><<<E / 32, 256, 0, stream>>>(
      edges, nullptr, nullptr, nullptr, nullptr, nullptr,
      P_encEW1, enc_edge_b1, P_encEW2, enc_edge_b2, h_e, nullptr, nullptr, nullptr, 128);

  // ---- 5 message-passing steps ----
  for (int i = 0; i < S; ++i) {
    long long zn = (long long)2 * N * 128;
    zero_f32<<<(int)((zn + 255) / 256), 256, 0, stream>>>(sent, zn);
    fused_mlp<384, 8, M_EDGE_STEP><<<E / 32, 256, 0, stream>>>(
        h_e, nullptr, nullptr, h_n16, senders, receivers,
        P_eW1 + (size_t)i * 8 * 12 * 512, eb1 + i * L,
        P_eW2 + (size_t)i * 8 * 4 * 512,  eb2 + i * L,
        h_e, nullptr, sent, recv, 128);
    fused_mlp<384, 8, M_NODE_STEP><<<N / 32, 256, 0, stream>>>(
        h_n, sent, recv, h_n16, nullptr, nullptr,
        P_nW1 + (size_t)i * 8 * 12 * 512, nb1 + i * L,
        P_nW2 + (size_t)i * 8 * 4 * 512,  nb2 + i * L,
        h_n, h_n16, nullptr, nullptr, 128);
  }

  // ---- decode ----
  float* out_n = (float*)d_out;          // [N, 3]
  float* out_e = out_n + (size_t)N * ON; // [E, 1]
  fused_mlp<128, 1, M_DEC_NODE><<<N / 32, 256, 0, stream>>>(
      h_n, nullptr, nullptr, nullptr, nullptr, nullptr,
      P_dNW1, dec_node_b1, P_dNW2, dec_node_b2, out_n, nullptr, nullptr, nullptr, ON);
  fused_mlp<128, 1, M_DEC_EDGE><<<E / 32, 256, 0, stream>>>(
      h_e, nullptr, nullptr, nullptr, nullptr, nullptr,
      P_dEW1, dec_edge_b1, P_dEW2, dec_edge_b2, out_e, nullptr, nullptr, nullptr, 1);
}
